// LSTMAggregator_4879082849100
// MI455X (gfx1250) — matrix operand order repaired and checked by reading, not compiled, not run on hardware
//
#include <hip/hip_runtime.h>
#include <hip/hip_bf16.h>

// ---------------------------------------------------------------------------
// LSTM aggregator, fused, bf16 WMMA (f32 accumulate) for gfx1250 / MI455X.
// B=16384, T=25, D=H=OUT=128, 4H=512.
// One 8-wave workgroup per 16 batch rows.
//  - Weights pre-packed (prologue kernel) to bf16 in d_ws in the exact
//    V_WMMA_F32_16X16X32_BF16 B-operand per-lane fragment layout
//    -> one pair of global b128 loads per fragment.
//  - Activations (x_t, h) staged in LDS as bf16 row-major; the 16-bit A
//    operand layout is two contiguous 8-element K runs per lane, so each
//    A fragment is exactly two ds_load_b128 (no per-fragment conversion).
//  - __launch_bounds__(256,4): cap at 256 VGPRs -> no s_set_vgpr_msb,
//    4 waves/SIMD occupancy for latency hiding.
// ---------------------------------------------------------------------------

typedef __bf16 bf16_t;
typedef bf16_t v16bf __attribute__((ext_vector_type(16)));
typedef float  v8f   __attribute__((ext_vector_type(8)));

#define B_TOT   16384
#define T_SEQ   25
#define D_IN    128
#define H_DIM   128
#define OUT_D   128
#define XSTRIDE 136              // bf16 units; 272 B = 68 words = 4 mod 64 banks

#define PK_GATE_ELEMS (8 * 32 * 32 * 16)   // [kc=8][ntile=32][lane=32][j=16] bf16
#define PK_OUT_ELEMS  (8 * 8 * 32 * 16)    // [kc=8][ntile=8 ][lane=32][j=16] bf16

__device__ __forceinline__ unsigned f2bf_rne(float f) {
  unsigned u = __float_as_uint(f);
  return (u + 0x7FFFu + ((u >> 16) & 1u)) >> 16;     // round-to-nearest-even
}
__device__ __forceinline__ unsigned pack2bf(float lo, float hi) {
  return f2bf_rne(lo) | (f2bf_rne(hi) << 16);
}

union ABFrag {
  unsigned u[8];
  uint4    q[2];
  v16bf    v;
};

__device__ __forceinline__ v8f wmma_bf16(v16bf a, v16bf b, v8f c) {
  return __builtin_amdgcn_wmma_f32_16x16x32_bf16(
      /*neg_a=*/false, a, /*neg_b=*/false, b,
      /*c_mod=*/(short)0, c, /*reuse_a=*/false, /*reuse_b=*/false);
}

// A fragment (16x32 bf16) from a bf16 LDS tile [16][XSTRIDE], row-major.
// Per-lane data = K runs [kbase+hi, +7] and [kbase+16+hi, +7] of row lane&15,
// which is exactly two aligned 16-byte LDS loads.
__device__ __forceinline__ v16bf build_a(const unsigned short* __restrict__ lds,
                                         int kbase, int lane) {
  ABFrag a;
  const int r  = lane & 15;
  const int hi = (lane >> 4) << 3;
  const unsigned short* p = lds + r * XSTRIDE + kbase + hi;
  a.q[0] = *(const uint4*)(p);
  a.q[1] = *(const uint4*)(p + 16);
  return a.v;
}

// Pre-packed 32x16 bf16 B-fragment: 32 contiguous bytes per lane.
__device__ __forceinline__ v16bf load_b(const unsigned short* __restrict__ pk,
                                        int cidx, int ntiles, int ntile, int lane) {
  ABFrag b;
  const uint4* p =
      (const uint4*)(pk + (((size_t)((cidx * ntiles + ntile) << 5) + lane) << 4));
  b.q[0] = p[0];
  b.q[1] = p[1];
  return b.v;
}

__device__ __forceinline__ float fsigmoid(float x) { return 1.0f / (1.0f + __expf(-x)); }
__device__ __forceinline__ float ftanh(float x)    { return 2.0f / (1.0f + __expf(-2.0f * x)) - 1.0f; }

// ---------------------------------------------------------------------------
// Prologue: pack [kernel;rec_kernel] (256x512) and [W_self;W_neigh] (256x128)
// into bf16 B-operand fragment order: per (kc, ntile, lane): 16 contiguous
// bf16 = K rows {kc*32 + (j>>3)*16 + (lane>>4)*8 + (j&7)}, column {ntile*16 + (lane&15)}.
// ---------------------------------------------------------------------------
__global__ void pack_weights(const float* __restrict__ kern,
                             const float* __restrict__ rec,
                             const float* __restrict__ wself,
                             const float* __restrict__ wneigh,
                             unsigned short* __restrict__ pk_gate,
                             unsigned short* __restrict__ pk_out) {
  int idx = blockIdx.x * 256 + threadIdx.x;
  if (idx < PK_GATE_ELEMS) {
    int j = idx & 15, lane = (idx >> 4) & 31, ntile = (idx >> 9) & 31, c = idx >> 14;
    int K = c * 32 + ((j >> 3) << 4) + ((lane >> 4) << 3) + (j & 7);
    int n = ntile * 16 + (lane & 15);
    float v = (K < 128) ? kern[K * 512 + n] : rec[(K - 128) * 512 + n];
    pk_gate[idx] = (unsigned short)f2bf_rne(v);
  } else if (idx < PK_GATE_ELEMS + PK_OUT_ELEMS) {
    int i2 = idx - PK_GATE_ELEMS;
    int j = i2 & 15, lane = (i2 >> 4) & 31, ntile = (i2 >> 9) & 7, c = i2 >> 12;
    int K = c * 32 + ((j >> 3) << 4) + ((lane >> 4) << 3) + (j & 7);
    int n = ntile * 16 + (lane & 15);
    float v = (K < 128) ? wself[K * 128 + n] : wneigh[(K - 128) * 128 + n];
    pk_out[i2] = (unsigned short)f2bf_rne(v);
  }
}

// ---------------------------------------------------------------------------
// Main fused kernel: 256 threads (8 wave32) per 16 batch rows.
// ---------------------------------------------------------------------------
__global__ __launch_bounds__(256, 4)
void lstm_agg_kernel(const float* __restrict__ self_vecs,
                     const float* __restrict__ neig,
                     const float* __restrict__ bias,
                     const unsigned short* __restrict__ pk_gate,
                     const unsigned short* __restrict__ pk_out,
                     float* __restrict__ out) {
  __shared__ __align__(16) unsigned short xstage[16 * XSTRIDE];  // bf16 bits
  __shared__ __align__(16) unsigned short hbuf[16 * XSTRIDE];    // bf16 bits
  __shared__ int usedf[2][16];

  const int tid  = threadIdx.x;
  const int lane = tid & 31;
  const int wid  = tid >> 5;
  const int b0   = blockIdx.x << 4;
  const int r    = lane & 15;
  const int hi8  = (lane >> 4) << 3;
  const int ncol = wid * 16 + r;       // this lane's h/out column

  for (int i = tid; i < 16 * XSTRIDE; i += 256) hbuf[i] = 0;
  if (tid < 16) { usedf[0][tid] = 0; usedf[1][tid] = 0; }

  float sb[4];
#pragma unroll
  for (int g = 0; g < 4; ++g) sb[g] = bias[g * H_DIM + ncol];

  v8f c_st, out_h;
#pragma unroll
  for (int k = 0; k < 8; ++k) { c_st[k] = 0.0f; out_h[k] = 0.0f; }

  for (int t = 0; t < T_SEQ; ++t) {
    __syncthreads();                           // prev h published, xstage free
    // stage x_t as bf16 pairs + per-row "used" flags (benign races);
    // reset the next step's flag buffer
#pragma unroll
    for (int i = 0; i < 4; ++i) {
      int idx = tid + i * 256;                 // 1024 float2 pairs
      int rr = idx >> 6, dp = idx & 63;        // row, pair index (d = 2*dp)
      const float2 xv = *(const float2*)(neig +
          (size_t)(b0 + rr) * (T_SEQ * D_IN) + (size_t)t * D_IN + dp * 2);
      *(unsigned*)&xstage[rr * XSTRIDE + dp * 2] = pack2bf(xv.x, xv.y);
      if (xv.x != 0.0f || xv.y != 0.0f) usedf[t & 1][rr] = 1;
    }
    if (tid < 16) usedf[(t + 1) & 1][tid] = 0;
    __syncthreads();                           // x staged, flags ready

    v8f acc[4];
#pragma unroll
    for (int g = 0; g < 4; ++g)
#pragma unroll
      for (int k = 0; k < 8; ++k) acc[g][k] = sb[g];

    // z = x_t @ kernel  (K = 0..127 -> packed kc 0..3)
#pragma unroll
    for (int kc = 0; kc < 4; ++kc) {
      v16bf a = build_a(xstage, kc * 32, lane);
#pragma unroll
      for (int g = 0; g < 4; ++g) {
        v16bf b = load_b(pk_gate, kc, 32, g * 8 + wid, lane);
        acc[g] = wmma_bf16(a, b, acc[g]);
      }
    }
    // z += h @ rec_kernel  (packed kc 4..7)
#pragma unroll
    for (int kc = 0; kc < 4; ++kc) {
      v16bf a = build_a(hbuf, kc * 32, lane);
#pragma unroll
      for (int g = 0; g < 4; ++g) {
        v16bf b = load_b(pk_gate, 4 + kc, 32, g * 8 + wid, lane);
        acc[g] = wmma_bf16(a, b, acc[g]);
      }
    }

    // LSTM cell (elementwise in C/D register layout)
    v8f h_new;
#pragma unroll
    for (int k = 0; k < 8; ++k) {
      float ig = fsigmoid(acc[0][k]);
      float fg = fsigmoid(acc[1][k]);
      float gg = ftanh(acc[2][k]);
      float og = fsigmoid(acc[3][k]);
      float cc = fg * c_st[k] + ig * gg;
      c_st[k]  = cc;
      h_new[k] = og * ftanh(cc);
    }

    __syncthreads();                           // all hbuf reads done
#pragma unroll
    for (int k = 0; k < 8; ++k) {
      int rrow = k + hi8;
      hbuf[rrow * XSTRIDE + ncol] = (unsigned short)f2bf_rne(h_new[k]);
      // capture h at (length-1): overwrite while the step is used; t==0 is
      // the floor for all-empty rows (length = max(sum(used),1)).
      if (t == 0 || usedf[t & 1][rrow]) out_h[k] = h_new[k];
    }
  }

  __syncthreads();
  // publish selected h (bf16) and stage self tile for the output GEMM
#pragma unroll
  for (int k = 0; k < 8; ++k)
    hbuf[(k + hi8) * XSTRIDE + ncol] = (unsigned short)f2bf_rne(out_h[k]);
#pragma unroll
  for (int i = 0; i < 4; ++i) {
    int idx = tid + i * 256;
    int rr = idx >> 6, dp = idx & 63;
    const float2 sv = *(const float2*)(self_vecs +
        (size_t)(b0 + rr) * D_IN + dp * 2);
    *(unsigned*)&xstage[rr * XSTRIDE + dp * 2] = pack2bf(sv.x, sv.y);
  }
  __syncthreads();

  // out = relu(self @ W_self + h_sel @ W_neigh); each wave owns one 16-col tile
  v8f acc;
#pragma unroll
  for (int k = 0; k < 8; ++k) acc[k] = 0.0f;
#pragma unroll
  for (int kc = 0; kc < 4; ++kc) {
    v16bf a = build_a(xstage, kc * 32, lane);
    v16bf b = load_b(pk_out, kc, 8, wid, lane);
    acc = wmma_bf16(a, b, acc);
  }
#pragma unroll
  for (int kc = 0; kc < 4; ++kc) {
    v16bf a = build_a(hbuf, kc * 32, lane);
    v16bf b = load_b(pk_out, 4 + kc, 8, wid, lane);
    acc = wmma_bf16(a, b, acc);
  }
#pragma unroll
  for (int k = 0; k < 8; ++k) {
    float v = acc[k];
    out[(size_t)(b0 + k + hi8) * OUT_D + ncol] = v > 0.0f ? v : 0.0f;
  }
}

extern "C" void kernel_launch(void* const* d_in, const int* in_sizes, int n_in,
                              void* d_out, int out_size, void* d_ws, size_t ws_size,
                              hipStream_t stream) {
  const float* self_vecs = (const float*)d_in[0];
  const float* neig      = (const float*)d_in[1];
  const float* kern      = (const float*)d_in[2];
  const float* rec       = (const float*)d_in[3];
  const float* bias      = (const float*)d_in[4];
  const float* wself     = (const float*)d_in[5];
  const float* wneigh    = (const float*)d_in[6];
  float* out = (float*)d_out;

  unsigned short* pk_gate = (unsigned short*)d_ws;            // 256 KB
  unsigned short* pk_out  = pk_gate + PK_GATE_ELEMS;          // +64 KB

  const int pack_total  = PK_GATE_ELEMS + PK_OUT_ELEMS;       // 163840
  const int pack_blocks = (pack_total + 255) / 256;           // 640
  pack_weights<<<pack_blocks, 256, 0, stream>>>(kern, rec, wself, wneigh,
                                                pk_gate, pk_out);

  lstm_agg_kernel<<<B_TOT / 16, 256, 0, stream>>>(self_vecs, neig, bias,
                                                  pk_gate, pk_out, out);
}
